// Dec_9285719294079
// MI455X (gfx1250) — matrix operand order repaired and checked by reading, not compiled, not run on hardware
//
#include <hip/hip_runtime.h>
#include <hip/hip_bf16.h>

// ---------------------------------------------------------------------------
// LSTM decoder (B=32, T=64, V=32000, D=512, E=1024, H=1024), teacher forcing.
// Strategy:
//   1) Convert W_ih / W_hh / W_fc to bf16 (stay resident in 192MB L2: ~87MB).
//   2) Build Xfull[T*B, E+D] bf16 = [encoder_b | emb[gtruths[b,t]]].
//   3) One big WMMA GEMM: gates_x = Xfull @ W_ih^T + (b_ih + b_hh).
//   4) Sequential t-loop: gates = gates_x[t] + h @ W_hh^T  (WMMA),
//      LSTM cell elementwise -> c (f32), h (bf16),
//      logits[t] = h @ W_fc^T + b_fc (WMMA, the dominant GEMM).
//   5) argmax over V -> predic.
// All matmuls use v_wmma_f32_16x16x32_bf16 (wave32, 16x16 tiles, K step 32).
// ---------------------------------------------------------------------------

#define B_ 32
#define T_ 64
#define V_ 32000
#define D_ 512
#define E_ 1024
#define H_ 1024
#define G4_ (4 * H_)      // 4096
#define ED_ (E_ + D_)     // 1536

typedef __bf16 bf16_t;
typedef __attribute__((ext_vector_type(16))) __bf16 v16bf;
typedef __attribute__((ext_vector_type(8)))  __bf16 v8bf;
typedef __attribute__((ext_vector_type(8)))  float  v8f;

// ---- A-tile loader: 16x32 bf16, per ISA 7.12.2 16-bit A layout ------------
// lanes 0-15 hold row M=lane, K = {k0..k0+7, k0+16..k0+23}
// lanes 16-31 hold row M=lane-16, K = {k0+8..k0+15, k0+24..k0+31}
__device__ __forceinline__ v16bf load_a_tile(const bf16_t* __restrict__ X, int ldx,
                                             int m0, int k0, int lane) {
    const int row  = m0 + (lane & 15);
    const int koff = (lane & 16) ? 8 : 0;
    const bf16_t* p = X + (size_t)row * ldx + k0 + koff;
    v8bf lo = *(const v8bf*)(p);
    v8bf hi = *(const v8bf*)(p + 16);
    return __builtin_shufflevector(lo, hi, 0, 1, 2, 3, 4, 5, 6, 7,
                                           8, 9, 10, 11, 12, 13, 14, 15);
}

// ---- B-tile loader: 32x16 bf16 (W stored row-major [N,K], i.e. B = W^T) ---
// lanes 0-15: col N=n0+lane, K = {k0..k0+7, k0+16..k0+23} ; lanes 16-31: K = {k0+8..k0+15, k0+24..k0+31}
__device__ __forceinline__ v16bf load_b_tile(const bf16_t* __restrict__ W, int ldw,
                                             int n0, int k0, int lane) {
    const int col  = n0 + (lane & 15);
    const int koff = (lane & 16) ? 8 : 0;
    const bf16_t* p = W + (size_t)col * ldw + k0 + koff;
    v8bf lo = *(const v8bf*)(p);
    v8bf hi = *(const v8bf*)(p + 16);
    return __builtin_shufflevector(lo, hi, 0, 1, 2, 3, 4, 5, 6, 7,
                                           8, 9, 10, 11, 12, 13, 14, 15);
}

// ---------------------------------------------------------------------------
// C[m0:m0+32, :N] = X[m0:m0+32, :K] @ W[:N, :K]^T  (+bias1+bias2)[col] (+addmat)
// One wave per 16-wide N strip; covers both M tiles to reuse the B operand.
// blockDim.x must be 256 (8 waves). N must be a multiple of 128.
// ---------------------------------------------------------------------------
__global__ void gemm32_bf16_wmma(const bf16_t* __restrict__ X, int ldx,
                                 const bf16_t* __restrict__ W, int ldw,
                                 float* __restrict__ C, int ldc,
                                 const float* __restrict__ bias1,
                                 const float* __restrict__ bias2,
                                 const float* __restrict__ addmat,
                                 int N, int K) {
    const int lane  = threadIdx.x & 31;
    const int wave  = threadIdx.x >> 5;
    const int n0    = (blockIdx.x * 8 + wave) * 16;
    const int m0    = blockIdx.y * 32;

    v8f acc0 = {};
    v8f acc1 = {};
    const bf16_t* prow = W + (size_t)(n0 + (lane & 15)) * ldw + ((lane & 16) ? 16 : 0);

    for (int k0 = 0; k0 < K; k0 += 32) {
        // pull the next K-slice of this weight row toward L0/L2
        __builtin_prefetch(prow + k0 + 128, 0, 0);
        v16bf b  = load_b_tile(W, ldw, n0, k0, lane);
        v16bf a0 = load_a_tile(X, ldx, m0,      k0, lane);
        v16bf a1 = load_a_tile(X, ldx, m0 + 16, k0, lane);
        acc0 = __builtin_amdgcn_wmma_f32_16x16x32_bf16(false, a0, false, b,
                                                       (short)0, acc0, false, false);
        acc1 = __builtin_amdgcn_wmma_f32_16x16x32_bf16(false, a1, false, b,
                                                       (short)0, acc1, false, false);
    }

    // C/D layout (ISA 7.12.2): VGPR i -> M = i (+8 for lanes 16-31), N = lane&15
    const int col = n0 + (lane & 15);
    const int rb  = (lane & 16) ? 8 : 0;
    float bv = 0.0f;
    if (bias1) bv += bias1[col];
    if (bias2) bv += bias2[col];
#pragma unroll
    for (int i = 0; i < 8; ++i) {
        const int r0 = m0 + rb + i;
        float v0 = acc0[i] + bv;
        float v1 = acc1[i] + bv;
        if (addmat) {
            v0 += addmat[(size_t)(rb + i) * ldc + col];
            v1 += addmat[(size_t)(rb + i + 16) * ldc + col];
        }
        C[(size_t)r0 * ldc + col]        = v0;
        C[(size_t)(r0 + 16) * ldc + col] = v1;
    }
}

// ---- fp32 -> bf16 conversion ---------------------------------------------
__global__ void f32_to_bf16_kernel(const float* __restrict__ src,
                                   bf16_t* __restrict__ dst, int n) {
    int i = blockIdx.x * blockDim.x + threadIdx.x;
    if (i < n) dst[i] = (bf16_t)src[i];
}

// ---- build Xfull[T*B, E+D] bf16 = [encoder | emb[gtruths]] ----------------
__global__ void build_x_kernel(const float* __restrict__ encoder,
                               const int* __restrict__ gtruths,
                               const float* __restrict__ emb,
                               bf16_t* __restrict__ Xf) {
    int idx = blockIdx.x * blockDim.x + threadIdx.x;
    if (idx >= T_ * B_ * ED_) return;
    int row = idx / ED_;
    int col = idx - row * ED_;
    int t = row >> 5, b = row & 31;
    float v;
    if (col < E_) {
        v = encoder[(size_t)b * E_ + col];
    } else {
        int tok = gtruths[b * T_ + t];
        v = emb[(size_t)tok * D_ + (col - E_)];
    }
    Xf[idx] = (bf16_t)v;
}

// ---- zero-init hidden state ----------------------------------------------
__global__ void init_state_kernel(bf16_t* __restrict__ hx, float* __restrict__ cx) {
    int i = blockIdx.x * blockDim.x + threadIdx.x;
    if (i < B_ * H_) { hx[i] = (bf16_t)0.0f; cx[i] = 0.0f; }
}

// ---- LSTM cell elementwise (torch gate order i,f,g,o) ---------------------
__global__ void lstm_cell_kernel(const float* __restrict__ gates, // [32, 4H]
                                 float* __restrict__ cx,          // [32, H]
                                 bf16_t* __restrict__ hx) {       // [32, H] bf16
    int idx = blockIdx.x * blockDim.x + threadIdx.x;
    if (idx >= B_ * H_) return;
    int b = idx >> 10, j = idx & (H_ - 1);
    const float* gr = gates + (size_t)b * G4_;
    float ig = gr[j];
    float fg = gr[H_ + j];
    float gg = gr[2 * H_ + j];
    float og = gr[3 * H_ + j];
    float si = 1.0f / (1.0f + __expf(-ig));
    float sf = 1.0f / (1.0f + __expf(-fg));
    float so = 1.0f / (1.0f + __expf(-og));
    float c  = sf * cx[idx] + si * tanhf(gg);
    float h  = so * tanhf(c);
    cx[idx] = c;
    hx[idx] = (bf16_t)h;
}

// ---- argmax over V per (t,b); predic[b,t] = argmax_v logits[t,b,v] --------
__global__ void argmax_kernel(const float* __restrict__ logits,
                              float* __restrict__ predic) {
    const int tb = blockIdx.x;              // t*32 + b
    const int t = tb >> 5, b = tb & 31;
    const float* row = logits + (size_t)tb * V_;
    __shared__ float smax[256];
    __shared__ int   sidx[256];
    float best = -3.402823466e38f;
    int bi = 0;
    for (int v = threadIdx.x; v < V_; v += 256) {
        float x = row[v];
        if (x > best) { best = x; bi = v; }
    }
    smax[threadIdx.x] = best;
    sidx[threadIdx.x] = bi;
    __syncthreads();
    for (int s = 128; s > 0; s >>= 1) {
        if (threadIdx.x < s) {
            float o = smax[threadIdx.x + s];
            int   oi = sidx[threadIdx.x + s];
            if (o > smax[threadIdx.x] ||
                (o == smax[threadIdx.x] && oi < sidx[threadIdx.x])) {
                smax[threadIdx.x] = o;
                sidx[threadIdx.x] = oi;
            }
        }
        __syncthreads();
    }
    if (threadIdx.x == 0) predic[b * T_ + t] = (float)sidx[0];
}

// ---------------------------------------------------------------------------
extern "C" void kernel_launch(void* const* d_in, const int* in_sizes, int n_in,
                              void* d_out, int out_size, void* d_ws, size_t ws_size,
                              hipStream_t stream) {
    const float* encoder = (const float*)d_in[0];
    const int*   gtruths = (const int*)d_in[1];
    const float* emb     = (const float*)d_in[2];
    const float* W_ih    = (const float*)d_in[3];
    const float* W_hh    = (const float*)d_in[4];
    const float* b_ih    = (const float*)d_in[5];
    const float* b_hh    = (const float*)d_in[6];
    const float* W_fc    = (const float*)d_in[7];
    const float* b_fc    = (const float*)d_in[8];

    float* logits = (float*)d_out;                         // [T, B, V] fp32
    float* predic = logits + (size_t)T_ * B_ * V_;         // [B, T] as float

    // workspace carve-out
    char*  ws = (char*)d_ws;
    size_t o  = 0;
    auto carve = [&](size_t bytes) -> void* {
        void* p = ws + o;
        o = (o + bytes + 255) & ~(size_t)255;
        return p;
    };
    bf16_t* Wih_b   = (bf16_t*)carve((size_t)G4_ * ED_ * 2);   // 12.6 MB
    bf16_t* Whh_b   = (bf16_t*)carve((size_t)G4_ * H_ * 2);    //  8.4 MB
    bf16_t* Wfc_b   = (bf16_t*)carve((size_t)V_ * H_ * 2);     // 65.5 MB
    bf16_t* Xfull   = (bf16_t*)carve((size_t)T_ * B_ * ED_ * 2); // 6.3 MB
    float*  gates_x = (float*)carve((size_t)T_ * B_ * G4_ * 4);  // 33.6 MB
    float*  gates   = (float*)carve((size_t)B_ * G4_ * 4);       // 0.5 MB
    bf16_t* hx      = (bf16_t*)carve((size_t)B_ * H_ * 2);
    float*  cx      = (float*)carve((size_t)B_ * H_ * 4);

    // 1) bf16 weight conversion (weights then live in L2 across all steps)
    {
        int n1 = G4_ * ED_, n2 = G4_ * H_, n3 = V_ * H_;
        f32_to_bf16_kernel<<<(n1 + 255) / 256, 256, 0, stream>>>(W_ih, Wih_b, n1);
        f32_to_bf16_kernel<<<(n2 + 255) / 256, 256, 0, stream>>>(W_hh, Whh_b, n2);
        f32_to_bf16_kernel<<<(n3 + 255) / 256, 256, 0, stream>>>(W_fc, Wfc_b, n3);
    }

    // 2) gather/concat inputs for all T steps (teacher forcing => known ahead)
    {
        int n = T_ * B_ * ED_;
        build_x_kernel<<<(n + 255) / 256, 256, 0, stream>>>(encoder, gtruths, emb, Xfull);
    }
    init_state_kernel<<<(B_ * H_ + 255) / 256, 256, 0, stream>>>(hx, cx);

    // 3) one big GEMM: gates_x[T*B, 4H] = Xfull @ W_ih^T + b_ih + b_hh
    {
        dim3 grid(G4_ / 16 / 8, (T_ * B_) / 32);   // 32 x 64
        gemm32_bf16_wmma<<<grid, 256, 0, stream>>>(
            Xfull, ED_, Wih_b, ED_, gates_x, G4_,
            b_ih, b_hh, nullptr, G4_, ED_);
    }

    // 4) sequential recurrence: only h@W_hh^T + cell + FC per step
    for (int t = 0; t < T_; ++t) {
        // gates = gates_x[t] + hx @ W_hh^T
        gemm32_bf16_wmma<<<dim3(G4_ / 16 / 8, 1), 256, 0, stream>>>(
            hx, H_, Whh_b, H_, gates, G4_,
            nullptr, nullptr, gates_x + (size_t)t * B_ * G4_, G4_, H_);

        lstm_cell_kernel<<<(B_ * H_ + 255) / 256, 256, 0, stream>>>(gates, cx, hx);

        // logits[t] = hx @ W_fc^T + b_fc   (dominant GEMM, W_fc from L2)
        gemm32_bf16_wmma<<<dim3(V_ / 16 / 8, 1), 256, 0, stream>>>(
            hx, H_, Wfc_b, H_, logits + (size_t)t * B_ * V_, V_,
            b_fc, nullptr, nullptr, V_, H_);
    }

    // 5) predic[b,t] = argmax_v logits[t,b,v]
    argmax_kernel<<<T_ * B_, 256, 0, stream>>>(logits, predic);
}
